// hbPass_69715909148837
// MI455X (gfx1250) — matrix operand order repaired and checked by reading, not compiled, not run on hardware
//
#include <hip/hip_runtime.h>

typedef __attribute__((ext_vector_type(16))) _Float16 v16h;
typedef __attribute__((ext_vector_type(8)))  _Float16 v8h;
typedef __attribute__((ext_vector_type(8)))  float    v8f;

#define HDIM 114
#define WDIM 114
#define CDIM 64
#define BDIM 8
#define NPIX (HDIM * WDIM)     // 12996
#define PW   116               // padded width (pad-1 halo folded into layout)
#define PPIX (PW * PW)         // 13456
#define KTOT (CDIM * 9)        // 576
#define KC   32                // K chunk per WMMA step
#define NCHUNK (KTOT / KC)     // 18
#define TILE_P 64              // pixels per workgroup

// ------- Kernel 0: conv_w fp32 -> f16 (flat [cout][cin*9]) + koff table -----
__global__ void wcvt_kernel(const float* __restrict__ w, _Float16* __restrict__ wf,
                            int* __restrict__ kofftab, int n) {
    int i = blockIdx.x * blockDim.x + threadIdx.x;
    if (i < n) wf[i] = (_Float16)w[i];
    if (i < KTOT) {
        int cin = i / 9;
        int r   = i - cin * 9;
        int kh  = r / 3;
        int kw  = r - kh * 3;
        kofftab[i] = cin * PPIX + kh * PW + kw;   // padded-space element offset
    }
}

// ---- Kernel 1: fused im2col -> bin_active -> col2im, into padded f16 y -----
// y[b,c,h,w] = sign(x) * ni(h)/4 * sum_{j in {0,1,2}, 0<=w-j<=111} T[j + 4*floor((w-j)/4)]
// where T[s] = sliding 4-sum of |x| within row h.  Written into [116][116] with
// a zero halo so the conv kernel needs no bounds checks.
__global__ void binfold_kernel(const float* __restrict__ x, _Float16* __restrict__ ypad) {
    int idx = blockIdx.x * blockDim.x + threadIdx.x;
    const int total = BDIM * CDIM * PPIX;
    if (idx >= total) return;
    int ww = idx % PW;
    int hh = (idx / PW) % PW;
    int bc = idx / PPIX;

    if (hh == 0 || hh == PW - 1 || ww == 0 || ww == PW - 1) {
        ypad[idx] = (_Float16)0.f;   // halo
        return;
    }
    int h = hh - 1, w = ww - 1;
    const float* row = x + (size_t)bc * NPIX + (size_t)h * WDIM;

    float v   = row[w];
    float sgn = (v > 0.f) ? 1.f : ((v < 0.f) ? -1.f : 0.f);
    int   ni  = (h < 2 ? h : 2) - (h > 111 ? h - 111 : 0) + 1;

    float acc = 0.f;
#pragma unroll
    for (int j = 0; j < 3; ++j) {
        int d = w - j;
        if (d < 0 || d > 111) continue;
        int s0 = j + (d & ~3);  // AGG block start; s0+3 <= 113 always
        acc += fabsf(row[s0]) + fabsf(row[s0 + 1]) + fabsf(row[s0 + 2]) + fabsf(row[s0 + 3]);
    }
    ypad[idx] = (_Float16)(sgn * (float)ni * 0.25f * acc);
}

// ---- Kernel 2: 3x3 pad-1 conv (64->64) as implicit GEMM via WMMA -----------
// Out[64, 12996] = W[64, 576] x im2col(Y)[576, 12996] per batch.
// WG: 256 thr = 8 waves; tile = 64 cout x 64 pixels; wave tile = 16x32 (A reused).
// Y staging: one pixel x 8 consecutive k per thread.  k-group is wave-uniform,
// so the 8 gather offsets come from one uniform s_load of the precomputed koff
// table; each gather is then 1 v_add + 1 global_load_d16, and the 8 halves
// store to LDS as a single ds_store_b128.
__global__ __launch_bounds__(256) void conv_wmma_kernel(
    const _Float16* __restrict__ ypad, const _Float16* __restrict__ wf,
    const int* __restrict__ kofftab, const float* __restrict__ bias,
    float* __restrict__ out) {

    __shared__ __align__(16) _Float16 Wlds[CDIM][KC];    // [cout][k]   4 KB
    __shared__ __align__(16) _Float16 Ylds[TILE_P][KC];  // [pixel][k]  4 KB

    const int tid      = threadIdx.x;
    const int lane     = tid & 31;
    const int wave     = tid >> 5;    // 0..7
    const int rowTile  = wave & 3;    // cout tile (0..3)
    const int colGroup = wave >> 2;   // 32-pixel group within WG (0..1)
    const int b        = blockIdx.y;
    const int pixBase  = blockIdx.x * TILE_P;

    // ---- weight staging role: 8 contiguous halves per thread per chunk ----
    const int wc_cout = tid >> 2;        // 0..63
    const int wc_kg   = (tid & 3) * 8;   // 0,8,16,24

    // ---- Y staging role: pixel = f(lane), k-group = f(wave) [wave-uniform] --
    const int kgs = __builtin_amdgcn_readfirstlane((tid >> 5) & 3);  // 0..3 scalar
    const int px  = ((tid >> 7) << 5) + lane;                        // 0..63
    // decompose (clamped) pixel ONCE; OOB pixels compute garbage columns that
    // are never stored (WMMA does not mix B columns).
    int P  = pixBase + px;
    int Pc = P < NPIX ? P : NPIX - 1;
    int ph = Pc / WDIM;
    int pw = Pc - ph * WDIM;
    const int off_px = ph * PW + pw;     // padded-space base (kh=kw=0 corner)

    const _Float16* ybat = ypad + (size_t)b * CDIM * PPIX;

    v8f acc0 = {}, acc1 = {};

#pragma unroll
    for (int kc = 0; kc < NCHUNK; ++kc) {
        const int k0 = kc * KC;
        __syncthreads();   // previous iteration's fragment reads complete

        // ---- stage weights (16B-aligned vector copy) ----
        *(v8h*)&Wlds[wc_cout][wc_kg] =
            *(const v8h*)&wf[(size_t)wc_cout * KTOT + k0 + wc_kg];
        if (kc + 1 < NCHUNK)  // prefetch next chunk's weight line
            __builtin_prefetch(&wf[(size_t)wc_cout * KTOT + k0 + KC + wc_kg], 0, 3);

        // ---- stage im2col gather: 8 consecutive k at one pixel ----
        {
            const int* kp = kofftab + kgs * 8 + k0;   // uniform -> s_load_b256
            v8h yv;
#pragma unroll
            for (int e = 0; e < 8; ++e) {
                int voff = off_px + kp[e];            // 1 v_add (SGPR + VGPR)
                yv[e] = ybat[voff];                   // global_load_d16, GVS form
            }
            *(v8h*)&Ylds[px][kgs * 8] = yv;           // one ds_store_b128
        }
        __syncthreads();

        // ---- A fragment (weights 16x32): lanes 0-15 K{0..15}, 16-31 K{16..31}
        const _Float16* wrow = &Wlds[rowTile * 16 + (lane & 15)][0];
        int baseA = (lane >> 4) << 4;   // 0 or 16
        v8h alo = *(const v8h*)(wrow + baseA);
        v8h ahi = *(const v8h*)(wrow + baseA + 8);
        v16h a = __builtin_shufflevector(alo, ahi, 0, 1, 2, 3, 4, 5, 6, 7,
                                         8, 9, 10, 11, 12, 13, 14, 15);

        // ---- two B fragments (ycol 32x16 each): lane half selects K 0-15 / 16-31
        int baseB = (lane >> 4) << 4;   // 0 or 16
        const _Float16* yrow0 = &Ylds[colGroup * 32 + (lane & 15)][0];
        v8h b0lo = *(const v8h*)(yrow0 + baseB);
        v8h b0hi = *(const v8h*)(yrow0 + baseB + 8);
        v16h bm0 = __builtin_shufflevector(b0lo, b0hi, 0, 1, 2, 3, 4, 5, 6, 7,
                                           8, 9, 10, 11, 12, 13, 14, 15);
        const _Float16* yrow1 = &Ylds[colGroup * 32 + 16 + (lane & 15)][0];
        v8h b1lo = *(const v8h*)(yrow1 + baseB);
        v8h b1hi = *(const v8h*)(yrow1 + baseB + 8);
        v16h bm1 = __builtin_shufflevector(b1lo, b1hi, 0, 1, 2, 3, 4, 5, 6, 7,
                                           8, 9, 10, 11, 12, 13, 14, 15);

        acc0 = __builtin_amdgcn_wmma_f32_16x16x32_f16(false, a, false, bm0,
                                                      (short)0, acc0, false, false);
        acc1 = __builtin_amdgcn_wmma_f32_16x16x32_f16(false, a, false, bm1,
                                                      (short)0, acc1, false, false);
    }

    // ---- epilogue: bias + ReLU; C/D layout: VGPR q -> M = q (+8 for lanes>=16)
    const int coutBase = rowTile * 16 + ((lane >> 4) << 3);
    const int P0 = pixBase + colGroup * 32 + (lane & 15);
    const int P1 = P0 + 16;
#pragma unroll
    for (int q = 0; q < 8; ++q) {
        int   cout = coutBase + q;
        float bv   = bias[cout];
        size_t ob  = (size_t)(b * CDIM + cout) * NPIX;
        if (P0 < NPIX) {
            float v = acc0[q] + bv;
            out[ob + P0] = v > 0.f ? v : 0.f;
        }
        if (P1 < NPIX) {
            float v = acc1[q] + bv;
            out[ob + P1] = v > 0.f ? v : 0.f;
        }
    }
}

extern "C" void kernel_launch(void* const* d_in, const int* in_sizes, int n_in,
                              void* d_out, int out_size, void* d_ws, size_t ws_size,
                              hipStream_t stream) {
    const float* x  = (const float*)d_in[0];  // [8,64,114,114]
    const float* cw = (const float*)d_in[1];  // [64,64,3,3]
    const float* cb = (const float*)d_in[2];  // [64]
    float* out = (float*)d_out;               // [8,64,114,114]

    // Workspace: [wf16 : 73728 B] [koff : 2304 B] [ypad : 8*64*116*116 halves]
    _Float16* wf16 = (_Float16*)d_ws;
    int*      koff = (int*)((char*)d_ws + 73728);
    _Float16* ypad = (_Float16*)((char*)d_ws + 73728 + 2304);

    {   // Kernel 0: weight conversion + koff table
        int n = CDIM * CDIM * 9;
        wcvt_kernel<<<(n + 255) / 256, 256, 0, stream>>>(cw, wf16, koff, n);
    }
    {   // Kernel 1: fused binarize + fold -> padded y (f16, zero halo)
        int total = BDIM * CDIM * PPIX;  // 6,889,472
        binfold_kernel<<<(total + 255) / 256, 256, 0, stream>>>(x, ypad);
    }
    {   // Kernel 2: WMMA implicit-GEMM conv + bias + ReLU
        dim3 grid((NPIX + TILE_P - 1) / TILE_P, BDIM);  // (204, 8)
        conv_wmma_kernel<<<grid, 256, 0, stream>>>(ypad, wf16, koff, cb, out);
    }
}
